// RecurrentGlobalModel_37177236914577
// MI455X (gfx1250) — compile-verified, matrix operand order checked by reading, not run on hardware
//
#include <hip/hip_runtime.h>
#include <hip/hip_bf16.h>

// ---------------------------------------------------------------------------
// Types for CDNA5 WMMA (wave32): A/B = 16 bf16 per lane, C/D = 8 f32 per lane
// ---------------------------------------------------------------------------
typedef __attribute__((ext_vector_type(16))) __bf16 v16bf;
typedef __attribute__((ext_vector_type(8)))  float  v8f;

#define NB   1024   // number of batch segments (B)
#define FEAT 128    // features of x / edge_attr / u / out
#define HS   256    // hidden size
#define INCZ 384    // concat input to fc1

__device__ inline __bf16 f2bf(float f) {
  unsigned u = __builtin_bit_cast(unsigned, f);
  unsigned r = u + 0x7FFFu + ((u >> 16) & 1u);   // round-to-nearest-even
  unsigned short s = (unsigned short)(r >> 16);
  return __builtin_bit_cast(__bf16, s);
}

// ---------------------------------------------------------------------------
// Prep kernels
// ---------------------------------------------------------------------------
__global__ void zero_f32(float* __restrict__ p, int n) {
  int i = blockIdx.x * blockDim.x + threadIdx.x;
  if (i < n) p[i] = 0.0f;
}

__global__ void cvt_bf16(const float* __restrict__ src, unsigned short* __restrict__ dst, int n) {
  int i = blockIdx.x * blockDim.x + threadIdx.x;
  if (i < n) dst[i] = __builtin_bit_cast(unsigned short, f2bf(src[i]));
}

// ---------------------------------------------------------------------------
// Scatter-sum with LDS privatization, tuned for HBM roofline:
//  - 16-feature slices -> 68 KB LDS/block -> 4 blocks (32 waves) per WGP
//  - float4 (b128) per-lane loads: 512 B per wave per load instruction
//  - 4 edges staged per iteration: all loads issued before any ds_add_f32,
//    keeping ~2 KB per wave in flight (value loads independent of seg gather)
//  - segment-swizzled LDS banks: slot = (f + s) & 15
// Block (bx, by): by = 16-feature slice, bx = item chunk.
// ---------------------------------------------------------------------------
__global__ void __launch_bounds__(256)
scatter_slice(const float* __restrict__ src,   // [nitems, 128]
              const int*   __restrict__ idx,   // edges: row ids; nodes: nullptr
              const int*   __restrict__ seg,   // batch[] over nodes
              float* __restrict__ sums,        // [1024, 128]
              float* __restrict__ cnts,        // [1024]
              int nitems)
{
  extern __shared__ float smem[];              // 1024*16 acc + 1024 cnt = 68 KB
  float* acc = smem;
  float* cnt = smem + NB * 16;

  const int  tid      = threadIdx.x;
  const int  featbase = blockIdx.y * 16;
  const bool do_count = (blockIdx.y == 0);

  for (int i = tid; i < NB * 17; i += 256) smem[i] = 0.0f;
  __syncthreads();

  const int per = (nitems + gridDim.x - 1) / gridDim.x;
  const int e0  = blockIdx.x * per;
  const int e1  = min(nitems, e0 + per);
  const int fl4 = (tid & 3) << 2;   // 4-feature group within the 16-wide slice
  const int sub = tid >> 2;         // 64 items in flight per block iteration
  const int ES  = 64;

  int e = e0 + sub;
  // Main quad-staged loop: maximize memory-level parallelism
  for (; e + 3 * ES < e1; e += 4 * ES) {
    int s[4];
    float4 v[4];
#pragma unroll
    for (int q = 0; q < 4; ++q) {
      int ee   = e + q * ES;
      int srow = idx ? idx[ee] : ee;
      s[q] = seg[srow];
      v[q] = *reinterpret_cast<const float4*>(src + (size_t)ee * FEAT + featbase + fl4);
    }
#pragma unroll
    for (int q = 0; q < 4; ++q) {
      float* base = &acc[s[q] * 16];
      atomicAdd(&base[(fl4 + 0 + s[q]) & 15], v[q].x);
      atomicAdd(&base[(fl4 + 1 + s[q]) & 15], v[q].y);
      atomicAdd(&base[(fl4 + 2 + s[q]) & 15], v[q].z);
      atomicAdd(&base[(fl4 + 3 + s[q]) & 15], v[q].w);
      if (do_count && (tid & 3) == 0) atomicAdd(&cnt[s[q]], 1.0f);
    }
  }
  // Tail
  for (; e < e1; e += ES) {
    int srow = idx ? idx[e] : e;
    int s    = seg[srow];
    float4 v = *reinterpret_cast<const float4*>(src + (size_t)e * FEAT + featbase + fl4);
    float* base = &acc[s * 16];
    atomicAdd(&base[(fl4 + 0 + s) & 15], v.x);
    atomicAdd(&base[(fl4 + 1 + s) & 15], v.y);
    atomicAdd(&base[(fl4 + 2 + s) & 15], v.z);
    atomicAdd(&base[(fl4 + 3 + s) & 15], v.w);
    if (do_count && (tid & 3) == 0) atomicAdd(&cnt[s], 1.0f);
  }
  __syncthreads();

  // Sparse global flush (total flush volume independent of slice width)
  for (int i = tid; i < NB * 16; i += 256) {
    int   s = i >> 4;
    int   f = i & 15;
    float v = acc[s * 16 + ((f + s) & 15)];
    if (v != 0.0f) atomicAdd(&sums[s * FEAT + featbase + f], v);
  }
  if (do_count) {
    for (int i = tid; i < NB; i += 256) {
      float v = cnt[i];
      if (v != 0.0f) atomicAdd(&cnts[i], v);
    }
  }
}

// ---------------------------------------------------------------------------
// WMMA fragment helpers (documented CDNA5 wave32 layouts, 16-bit 16x32 tiles)
// ---------------------------------------------------------------------------
__device__ inline v16bf frag_A(const __bf16* __restrict__ base, int ld, int lane, int k0) {
  const int m = lane & 15, h = lane >> 4;
  v16bf a;
#pragma unroll
  for (int vv = 0; vv < 8; ++vv) {
    int kb = k0 + ((vv & 3) << 1) + (h << 3) + ((vv >> 2) << 4);
    a[2 * vv]     = base[m * ld + kb];
    a[2 * vv + 1] = base[m * ld + kb + 1];
  }
  return a;
}

// B[k][n] tile taken from PyTorch-layout weight W[out][in] (so B = W^T slice)
__device__ inline v16bf frag_BW(const __bf16* __restrict__ W, int ldk, int lane, int n0, int k0) {
  const int n = lane & 15, h = lane >> 4;
  v16bf b;
#pragma unroll
  for (int vv = 0; vv < 8; ++vv) {
    int kb = k0 + ((vv & 3) << 1) + (h << 3) + ((vv >> 2) << 4);
    b[2 * vv]     = W[(size_t)(n0 + n) * ldk + kb];
    b[2 * vv + 1] = W[(size_t)(n0 + n) * ldk + kb + 1];
  }
  return b;
}

__device__ inline v8f splat8(float x) {
  v8f c;
#pragma unroll
  for (int i = 0; i < 8; ++i) c[i] = x;
  return c;
}

__device__ inline v8f gemm_tile(const __bf16* __restrict__ A, int lda,
                                const __bf16* __restrict__ W, int ldk,
                                int n0, int K, int lane, v8f c) {
#pragma unroll 4
  for (int k0 = 0; k0 < K; k0 += 32) {
    v16bf a = frag_A(A, lda, lane, k0);
    v16bf b = frag_BW(W, ldk, lane, n0, k0);
    c = __builtin_amdgcn_wmma_f32_16x16x32_bf16(false, a, false, b, (short)0, c, false, false);
  }
  return c;
}

// ---------------------------------------------------------------------------
// Fused head: concat -> fc1 -> LayerNorm -> ReLU -> GRU cell -> fc2
// One block = 16 batch rows, 128 threads = 4 wave32s, each wave owns 64 cols.
// ---------------------------------------------------------------------------
__global__ void __launch_bounds__(128)
fused_head(const float* __restrict__ u,     // [1024,128]
           const float* __restrict__ hin,   // [1024,256]
           const float* __restrict__ aggn, const float* __restrict__ cntn,
           const float* __restrict__ agge, const float* __restrict__ cnte,
           const unsigned short* __restrict__ fc1w_u, const float* __restrict__ fc1b,
           const float* __restrict__ lng,  const float* __restrict__ lnb,
           const unsigned short* __restrict__ wih_u,  const unsigned short* __restrict__ whh_u,
           const float* __restrict__ bih,  const float* __restrict__ bhh,
           const unsigned short* __restrict__ fc2w_u, const float* __restrict__ fc2b,
           float* __restrict__ out,         // [1024,128]
           float* __restrict__ hout)        // [1024,256]
{
  const __bf16* fc1w = (const __bf16*)fc1w_u;
  const __bf16* wih  = (const __bf16*)wih_u;
  const __bf16* whh  = (const __bf16*)whh_u;
  const __bf16* fc2w = (const __bf16*)fc2w_u;

  __shared__ __bf16 zin [16][INCZ];   // 12 KB  concat input, bf16
  __shared__ __bf16 hbf [16][HS];     //  8 KB  h in bf16
  __shared__ float  zf  [16][HS];     // 16 KB  fc1 output f32 (for LN)
  __shared__ __bf16 zact[16][HS];     //  8 KB  LN+ReLU output
  __shared__ __bf16 hnew[16][HS];     //  8 KB  h_new bf16 for fc2
  __shared__ float  mu[16], rs[16], rcn[16], rce[16];

  const int tid  = threadIdx.x;
  const int lane = tid & 31;
  const int wv   = tid >> 5;
  const int row0 = blockIdx.x * 16;

  if (tid < 16) {
    rcn[tid] = 1.0f / fmaxf(cntn[row0 + tid], 1.0f);
    rce[tid] = 1.0f / fmaxf(cnte[row0 + tid], 1.0f);
  }
  __syncthreads();

  // Stage z_in = [u | agg_n_mean | agg_e_mean] and h (bf16)
  for (int i = tid; i < 16 * INCZ; i += 128) {
    int r = i / INCZ, c = i % INCZ;
    int row = row0 + r;
    float v;
    if (c < 128)       v = u[row * 128 + c];
    else if (c < 256)  v = aggn[row * 128 + (c - 128)] * rcn[r];
    else               v = agge[row * 128 + (c - 256)] * rce[r];
    zin[r][c] = f2bf(v);
  }
  for (int i = tid; i < 16 * HS; i += 128) {
    int r = i / HS, c = i % HS;
    hbf[r][c] = f2bf(hin[(size_t)(row0 + r) * HS + c]);
  }
  __syncthreads();

  // fc1: [16,384] x [384,256]
  {
    const int n = lane & 15, h2 = lane >> 4;
    for (int t = 0; t < 4; ++t) {
      int n0 = wv * 64 + t * 16;
      v8f c = splat8(fc1b[n0 + n]);
      c = gemm_tile(&zin[0][0], INCZ, fc1w, INCZ, n0, INCZ, lane, c);
#pragma unroll
      for (int r = 0; r < 8; ++r) zf[r + 8 * h2][n0 + n] = c[r];
    }
  }
  __syncthreads();

  // LayerNorm stats per row
  if (tid < 16) {
    float s = 0.0f, s2 = 0.0f;
    for (int c = 0; c < HS; ++c) { float v = zf[tid][c]; s += v; s2 += v * v; }
    float m = s * (1.0f / HS);
    mu[tid] = m;
    rs[tid] = rsqrtf(fmaxf(s2 * (1.0f / HS) - m * m, 0.0f) + 1e-5f);
  }
  __syncthreads();

  for (int i = tid; i < 16 * HS; i += 128) {
    int r = i >> 8, c = i & 255;
    float v = (zf[r][c] - mu[r]) * rs[r] * lng[c] + lnb[c];
    zact[r][c] = f2bf(fmaxf(v, 0.0f));
  }
  __syncthreads();

  // GRU cell: gi = zact @ w_ih^T + b_ih, gh = h @ w_hh^T + b_hh (gate order r,z,n)
  {
    const int n = lane & 15, h2 = lane >> 4;
    for (int t = 0; t < 4; ++t) {
      int j0 = wv * 64 + t * 16;      // output column tile in [0,256)
      v8f gr = splat8(bih[j0 + n]);
      gr = gemm_tile(&zact[0][0], HS, wih, HS, j0, HS, lane, gr);
      v8f hr = splat8(bhh[j0 + n]);
      hr = gemm_tile(&hbf[0][0],  HS, whh, HS, j0, HS, lane, hr);
      v8f gz = splat8(bih[256 + j0 + n]);
      gz = gemm_tile(&zact[0][0], HS, wih, HS, 256 + j0, HS, lane, gz);
      v8f hz = splat8(bhh[256 + j0 + n]);
      hz = gemm_tile(&hbf[0][0],  HS, whh, HS, 256 + j0, HS, lane, hz);
      v8f gn = splat8(bih[512 + j0 + n]);
      gn = gemm_tile(&zact[0][0], HS, wih, HS, 512 + j0, HS, lane, gn);
      v8f hn = splat8(bhh[512 + j0 + n]);
      hn = gemm_tile(&hbf[0][0],  HS, whh, HS, 512 + j0, HS, lane, hn);
#pragma unroll
      for (int r = 0; r < 8; ++r) {
        int rowi = r + 8 * h2;
        int row  = row0 + rowi;
        int col  = j0 + n;
        float rg = 1.0f / (1.0f + __expf(-(gr[r] + hr[r])));
        float zg = 1.0f / (1.0f + __expf(-(gz[r] + hz[r])));
        float ng = tanhf(gn[r] + rg * hn[r]);
        float ho = hin[(size_t)row * HS + col];
        float hv = (1.0f - zg) * ng + zg * ho;
        hnew[rowi][col] = f2bf(hv);
        hout[(size_t)row * HS + col] = hv;
      }
    }
  }
  __syncthreads();

  // fc2: out = h_new @ fc2_w^T + fc2_b
  {
    const int n = lane & 15, h2 = lane >> 4;
    for (int t = 0; t < 2; ++t) {
      int n0 = (wv * 2 + t) * 16;
      v8f c = splat8(fc2b[n0 + n]);
      c = gemm_tile(&hnew[0][0], HS, fc2w, HS, n0, HS, lane, c);
#pragma unroll
      for (int r = 0; r < 8; ++r)
        out[(size_t)(row0 + r + 8 * h2) * 128 + n0 + n] = c[r];
    }
  }
}

// ---------------------------------------------------------------------------
// Launch
// ---------------------------------------------------------------------------
extern "C" void kernel_launch(void* const* d_in, const int* in_sizes, int n_in,
                              void* d_out, int out_size, void* d_ws, size_t ws_size,
                              hipStream_t stream) {
  const float* x     = (const float*)d_in[0];
  const int*   eidx  = (const int*)  d_in[1];   // [2,E], row = first E
  const float* eattr = (const float*)d_in[2];
  const float* h     = (const float*)d_in[3];
  const float* u     = (const float*)d_in[4];
  const int*   batch = (const int*)  d_in[5];
  const float* fc1_w = (const float*)d_in[6];
  const float* fc1_b = (const float*)d_in[7];
  const float* ln_g  = (const float*)d_in[8];
  const float* ln_b  = (const float*)d_in[9];
  const float* w_ih  = (const float*)d_in[10];
  const float* w_hh  = (const float*)d_in[11];
  const float* b_ih  = (const float*)d_in[12];
  const float* b_hh  = (const float*)d_in[13];
  const float* fc2_w = (const float*)d_in[14];
  const float* fc2_b = (const float*)d_in[15];

  const int N = in_sizes[0] / FEAT;
  const int E = in_sizes[1] / 2;

  // Workspace layout
  float* aggn = (float*)d_ws;                 // [1024,128]
  float* agge = aggn + NB * FEAT;             // [1024,128]
  float* cntn = agge + NB * FEAT;             // [1024]
  float* cnte = cntn + NB;                    // [1024]
  unsigned short* fc1w_bf = (unsigned short*)(cnte + NB);
  unsigned short* wih_bf  = fc1w_bf + 256 * 384;
  unsigned short* whh_bf  = wih_bf  + 768 * 256;
  unsigned short* fc2w_bf = whh_bf  + 768 * 256;

  // 1) zero accumulators
  int nz = 2 * NB * FEAT + 2 * NB;
  zero_f32<<<(nz + 255) / 256, 256, 0, stream>>>(aggn, nz);

  // 2) bf16 weight conversion (stays hot in L2)
  cvt_bf16<<<(256 * 384 + 255) / 256, 256, 0, stream>>>(fc1_w, fc1w_bf, 256 * 384);
  cvt_bf16<<<(768 * 256 + 255) / 256, 256, 0, stream>>>(w_ih, wih_bf, 768 * 256);
  cvt_bf16<<<(768 * 256 + 255) / 256, 256, 0, stream>>>(w_hh, whh_bf, 768 * 256);
  cvt_bf16<<<(128 * 256 + 255) / 256, 256, 0, stream>>>(fc2_w, fc2w_bf, 128 * 256);

  // 3) scatter-sum + counts (LDS-privatized; 68 KB dynamic LDS -> 4 blocks/WGP)
  size_t shbytes = (size_t)(NB * 17) * sizeof(float);
  scatter_slice<<<dim3(128, 8), 256, shbytes, stream>>>(eattr, eidx, batch, agge, cnte, E);
  scatter_slice<<<dim3(32, 8), 256, shbytes, stream>>>(x, nullptr, batch, aggn, cntn, N);

  // 4) fused MLP + LayerNorm + GRU + head via bf16 WMMA
  float* outp = (float*)d_out;                // [1024,128]
  float* hout = outp + NB * FEAT;             // [1024,256]
  fused_head<<<64, 128, 0, stream>>>(u, h, aggn, cntn, agge, cnte,
                                     fc1w_bf, fc1_b, ln_g, ln_b,
                                     wih_bf, whh_bf, b_ih, b_hh,
                                     fc2w_bf, fc2_b, outp, hout);
}
